// MoE_73108933313034
// MI455X (gfx1250) — matrix operand order repaired and checked by reading, not compiled, not run on hardware
//
#include <hip/hip_runtime.h>
#include <stdint.h>

#define S_TOK   8192
#define DMODEL  1024
#define FFN     4096
#define NEXP    8
#define CAP     1024   // max(int(1.0*8192/8), 4)

typedef __attribute__((ext_vector_type(16))) __bf16       v16bf;
typedef __attribute__((ext_vector_type(8)))  float        v8f;
typedef __attribute__((ext_vector_type(4))) unsigned int  u32x4;
typedef __attribute__((ext_vector_type(2))) unsigned int  u32x2;
typedef __attribute__((ext_vector_type(4)))  float        f32x4;

__device__ __forceinline__ unsigned short f32_bf16(float f) {
  unsigned int u = __float_as_uint(f);
  u += 0x7FFFu + ((u >> 16) & 1u);          // round-to-nearest-even
  return (unsigned short)(u >> 16);
}

union Frag { v16bf v; u32x4 q[2]; };

__device__ __forceinline__ void wait_async0() {
#if defined(__has_builtin) && __has_builtin(__builtin_amdgcn_s_wait_asynccnt)
  __builtin_amdgcn_s_wait_asynccnt(0);
#else
  asm volatile("s_wait_asynccnt 0" ::: "memory");
#endif
}

// Async-copy 32B (two b128 transfers) of an A-tile row chunk: global bf16 -> LDS.
// ISA: INST_OFFSET is added to BOTH the LDS and the global address, so the
// second chunk is the same vdst/vaddr with offset:16. Tracked by ASYNCcnt.
__device__ __forceinline__ void issueA32B(const unsigned short* gp, const unsigned short* lp) {
  const unsigned       lds = (unsigned)(uintptr_t)lp;        // low 32 bits = LDS offset
  const unsigned long long ga = (unsigned long long)(uintptr_t)gp;
  asm volatile("global_load_async_to_lds_b128 %0, %1, off"           :: "v"(lds), "v"(ga) : "memory");
  asm volatile("global_load_async_to_lds_b128 %0, %1, off offset:16" :: "v"(lds), "v"(ga) : "memory");
}

// ---------------------------------------------------------------- gating ----
__global__ __launch_bounds__(256) void moe_gate(
    const float* __restrict__ x, const float* __restrict__ wg,
    int* __restrict__ tok_expert, float* __restrict__ tok_gate,
    float* __restrict__ me_part)
{
  __shared__ float sm[8][NEXP];
  const int tid  = threadIdx.x;
  const int wave = tid >> 5, lane = tid & 31;
  const int s = blockIdx.x * 8 + wave;

  float acc[NEXP];
  #pragma unroll
  for (int e = 0; e < NEXP; ++e) acc[e] = 0.f;

  const float* xr = x + (size_t)s * DMODEL;
  #pragma unroll 4
  for (int i = 0; i < DMODEL / 32; ++i) {
    const int d = i * 32 + lane;
    const float xv = xr[d];
    const f32x4 w0 = *(const f32x4*)(wg + (size_t)d * NEXP);
    const f32x4 w1 = *(const f32x4*)(wg + (size_t)d * NEXP + 4);
    acc[0] += xv * w0.x; acc[1] += xv * w0.y; acc[2] += xv * w0.z; acc[3] += xv * w0.w;
    acc[4] += xv * w1.x; acc[5] += xv * w1.y; acc[6] += xv * w1.z; acc[7] += xv * w1.w;
  }
  #pragma unroll
  for (int off = 16; off > 0; off >>= 1) {
    #pragma unroll
    for (int e = 0; e < NEXP; ++e) acc[e] += __shfl_xor(acc[e], off, 32);
  }

  if (lane == 0) {
    float mx = acc[0]; int arg = 0;
    #pragma unroll
    for (int e = 1; e < NEXP; ++e) if (acc[e] > mx) { mx = acc[e]; arg = e; }
    float p[NEXP], sum = 0.f;
    #pragma unroll
    for (int e = 0; e < NEXP; ++e) { p[e] = __expf(acc[e] - mx); sum += p[e]; }
    const float inv = 1.f / sum;
    #pragma unroll
    for (int e = 0; e < NEXP; ++e) sm[wave][e] = p[e] * inv;
    tok_expert[s] = arg;
    tok_gate[s]   = p[arg] * inv;
  }
  __syncthreads();
  if (tid < NEXP) {
    float t = 0.f;
    #pragma unroll
    for (int w = 0; w < 8; ++w) t += sm[w][tid];
    me_part[(size_t)blockIdx.x * NEXP + tid] = t;
  }
}

// ------------------------------------------------------- order-exact scan ----
__global__ __launch_bounds__(256) void moe_scan(
    const int* __restrict__ tok_expert, int* __restrict__ tok_slot,
    int* __restrict__ slot2tok, const float* __restrict__ me_part,
    float* __restrict__ out_tail)
{
  __shared__ float s_cnt[NEXP];
  __shared__ float s_me[NEXP];
  const int tid  = threadIdx.x;
  const int wave = tid >> 5, lane = tid & 31;

  for (int i = tid; i < NEXP * CAP; i += 256) slot2tok[i] = -1;
  __syncthreads();

  int base = 0;
  for (int t0 = 0; t0 < S_TOK; t0 += 32) {
    const int s = t0 + lane;
    const bool m = (tok_expert[s] == wave);
    const unsigned int bal = (unsigned int)__ballot(m);
    if (m) {
      const int slot = base + __popc(bal & ((1u << lane) - 1u));
      if (slot < CAP) { tok_slot[s] = slot; slot2tok[wave * CAP + slot] = s; }
      else            { tok_slot[s] = -1; }
    }
    base += __popc(bal);
  }
  float me = 0.f;
  for (int b = lane; b < S_TOK / 8; b += 32) me += me_part[(size_t)b * NEXP + wave];
  #pragma unroll
  for (int off = 16; off > 0; off >>= 1) me += __shfl_xor(me, off, 32);
  if (lane == 0) { s_cnt[wave] = (float)base; s_me[wave] = me; }
  __syncthreads();
  if (tid == 0) {
    float laux = 0.f;
    #pragma unroll
    for (int e = 0; e < NEXP; ++e)
      laux += (s_me[e] / (float)S_TOK) * (s_cnt[e] / (float)S_TOK);
    out_tail[0] = laux * (float)NEXP;
    #pragma unroll
    for (int e = 0; e < NEXP; ++e) out_tail[1 + e] = s_cnt[e];
  }
}

// ------------------------------------------------------------- dispatch -----
__global__ __launch_bounds__(256) void moe_dispatch(
    const float* __restrict__ x, const int* __restrict__ slot2tok,
    unsigned short* __restrict__ disp)
{
  const int row = blockIdx.x;           // e*CAP + c
  const int tid = threadIdx.x;
  const int t = slot2tok[row];
  u32x2 out = {0u, 0u};
  if (t >= 0) {
    const f32x4 v = *(const f32x4*)(x + (size_t)t * DMODEL + tid * 4);
    out.x = (unsigned int)f32_bf16(v.x) | ((unsigned int)f32_bf16(v.y) << 16);
    out.y = (unsigned int)f32_bf16(v.z) | ((unsigned int)f32_bf16(v.w) << 16);
  }
  *(u32x2*)(disp + (size_t)row * DMODEL + tid * 4) = out;
}

// -------------------------------------------- zero rows of dropped tokens ---
__global__ __launch_bounds__(256) void moe_zero_dropped(
    const int* __restrict__ tok_slot, float* __restrict__ out)
{
  const int s = blockIdx.x;
  if (tok_slot[s] >= 0) return;
  f32x4 z = {0.f, 0.f, 0.f, 0.f};
  *(f32x4*)(out + (size_t)s * DMODEL + threadIdx.x * 4) = z;
}

// ----------------------------------------------------- WMMA expert GEMM -----
// MODE 0: H = relu(A*B + bias) -> bf16 Hout        (GEMM1)
// MODE 1: out[token] = gate * (A*B + bias), scattered by slot2tok (GEMM2 fused)
// Double-buffered LDS; A tile via global_load_async_to_lds_b128 (ASYNCcnt),
// B tile via VGPRs with fp32->bf16 conversion, stored K-transposed.
template<int MODE>
__global__ __launch_bounds__(256) void moe_gemm(
    const unsigned short* __restrict__ A, const float* __restrict__ B,
    const float* __restrict__ bias, unsigned short* __restrict__ Hout,
    const int* __restrict__ slot2tok, const float* __restrict__ tok_gate,
    float* __restrict__ out, int M, int N, int K)
{
  constexpr int BM = 128, BN = 64, BK = 32;
  __shared__ unsigned short As[2][BM * BK];   // row-major [M][K], 2 x 8 KB
  __shared__ unsigned short Bs[2][BN * BK];   // K-transposed [N][K], 2 x 4 KB

  const int e  = blockIdx.z;
  const int n0 = blockIdx.x * BN;
  const int m0 = blockIdx.y * BM;
  const int tid  = threadIdx.x;
  const int lane = tid & 31, wave = tid >> 5;
  const int wM = (wave & 3) * 32;
  const int wN = (wave >> 2) * 32;
  const int half = lane >> 4;
  const int l16  = lane & 15;

  const unsigned short* Ae = A + (size_t)e * M * K;
  const float*          Be = B + (size_t)e * K * N;

  v8f acc[2][2];
  #pragma unroll
  for (int i = 0; i < 2; ++i)
    #pragma unroll
    for (int j = 0; j < 2; ++j)
      #pragma unroll
      for (int r = 0; r < 8; ++r) acc[i][j][r] = 0.f;

  const int arow = tid >> 1;            // 0..127
  const int acol = (tid & 1) * 16;      // 0,16
  const int brow = tid >> 3;            // 0..31 (K)
  const int bcol = (tid & 7) * 8;       // 0..56 (N)
  const unsigned short* Ag = Ae + (size_t)(m0 + arow) * K + acol;
  const float*          Bg = Be + (size_t)brow * N + (n0 + bcol);

  // ---- prologue: tile 0 ----
  issueA32B(Ag, &As[0][arow * BK + acol]);
  f32x4 f0 = *(const f32x4*)(Bg);
  f32x4 f1 = *(const f32x4*)(Bg + 4);
  {
    unsigned short* bs = &Bs[0][0];
    bs[(bcol + 0) * BK + brow] = f32_bf16(f0.x);
    bs[(bcol + 1) * BK + brow] = f32_bf16(f0.y);
    bs[(bcol + 2) * BK + brow] = f32_bf16(f0.z);
    bs[(bcol + 3) * BK + brow] = f32_bf16(f0.w);
    bs[(bcol + 4) * BK + brow] = f32_bf16(f1.x);
    bs[(bcol + 5) * BK + brow] = f32_bf16(f1.y);
    bs[(bcol + 6) * BK + brow] = f32_bf16(f1.z);
    bs[(bcol + 7) * BK + brow] = f32_bf16(f1.w);
  }
  wait_async0();
  __syncthreads();

  int buf = 0;
  for (int k0 = 0; k0 < K; k0 += BK, buf ^= 1) {
    const bool has_next = (k0 + BK) < K;
    f32x4 nf0, nf1;
    if (has_next) {
      issueA32B(Ag + (k0 + BK), &As[buf ^ 1][arow * BK + acol]);
      const float* bg = Bg + (size_t)(k0 + BK) * N;
      nf0 = *(const f32x4*)(bg);
      nf1 = *(const f32x4*)(bg + 4);
    }

    // fragments from current buffer
    Frag fa[2], fb[2];
    #pragma unroll
    for (int i = 0; i < 2; ++i) {   // A 16x32: lane=row, K chunks at 8*half, 16+8*half
      const unsigned short* ap = &As[buf][(wM + i * 16 + l16) * BK];
      fa[i].q[0] = *(const u32x4*)(ap + half * 8);
      fa[i].q[1] = *(const u32x4*)(ap + 16 + half * 8);
    }
    #pragma unroll
    for (int j = 0; j < 2; ++j) {   // B 32x16: lane=col, K chunks at 8*half, 16+8*half
      const unsigned short* bp = &Bs[buf][(wN + j * 16 + l16) * BK];
      fb[j].q[0] = *(const u32x4*)(bp + half * 8);
      fb[j].q[1] = *(const u32x4*)(bp + 16 + half * 8);
    }
    #pragma unroll
    for (int i = 0; i < 2; ++i)
      #pragma unroll
      for (int j = 0; j < 2; ++j)
        acc[i][j] = __builtin_amdgcn_wmma_f32_16x16x32_bf16(
            false, fa[i].v, false, fb[j].v, (short)0, acc[i][j], false, false);

    if (has_next) {
      unsigned short* bs = &Bs[buf ^ 1][0];
      bs[(bcol + 0) * BK + brow] = f32_bf16(nf0.x);
      bs[(bcol + 1) * BK + brow] = f32_bf16(nf0.y);
      bs[(bcol + 2) * BK + brow] = f32_bf16(nf0.z);
      bs[(bcol + 3) * BK + brow] = f32_bf16(nf0.w);
      bs[(bcol + 4) * BK + brow] = f32_bf16(nf1.x);
      bs[(bcol + 5) * BK + brow] = f32_bf16(nf1.y);
      bs[(bcol + 6) * BK + brow] = f32_bf16(nf1.z);
      bs[(bcol + 7) * BK + brow] = f32_bf16(nf1.w);
      wait_async0();          // next A tile fully in LDS
      __syncthreads();        // next B tile visible; old buffer free
    }
  }

  // epilogue: C/D layout N = lane%16, M = r + 8*(lane/16)
  #pragma unroll
  for (int j = 0; j < 2; ++j) {
    const int col = n0 + wN + j * 16 + l16;
    const float bvv = bias[(size_t)e * N + col];
    #pragma unroll
    for (int i = 0; i < 2; ++i) {
      const int rowBase = m0 + wM + i * 16 + half * 8;
      #pragma unroll
      for (int r = 0; r < 8; ++r) {
        const int row = rowBase + r;
        const float v = acc[i][j][r] + bvv;
        if constexpr (MODE == 0) {
          Hout[((size_t)e * M + row) * N + col] = f32_bf16(fmaxf(v, 0.f));
        } else {
          const int t = slot2tok[e * CAP + row];
          if (t >= 0) out[(size_t)t * DMODEL + col] = tok_gate[t] * v;
        }
      }
    }
  }
}

// ---------------------------------------------------------------- launch ----
extern "C" void kernel_launch(void* const* d_in, const int* in_sizes, int n_in,
                              void* d_out, int out_size, void* d_ws, size_t ws_size,
                              hipStream_t stream)
{
  const float* x  = (const float*)d_in[0];
  const float* wg = (const float*)d_in[1];
  const float* w1 = (const float*)d_in[2];
  const float* b1 = (const float*)d_in[3];
  const float* w2 = (const float*)d_in[4];
  const float* b2 = (const float*)d_in[5];
  float* out = (float*)d_out;

  char* W = (char*)d_ws;
  float*          me_part  = (float*)(W + 0);                       // 32 KB
  int*            tok_exp  = (int*)  (W + (64ull  << 10));          // 32 KB
  int*            tok_slot = (int*)  (W + (96ull  << 10));          // 32 KB
  float*          tok_gate = (float*)(W + (128ull << 10));          // 32 KB
  int*            slot2tok = (int*)  (W + (160ull << 10));          // 32 KB
  unsigned short* disp     = (unsigned short*)(W + (1ull  << 20));  // 16 MB  [E][C][D] bf16
  unsigned short* h        = (unsigned short*)(W + (17ull << 20));  // 64 MB  [E][C][F] bf16

  moe_gate    <<<S_TOK / 8, 256, 0, stream>>>(x, wg, tok_exp, tok_gate, me_part);
  moe_scan    <<<1, 256, 0, stream>>>(tok_exp, tok_slot, slot2tok, me_part,
                                      out + (size_t)S_TOK * DMODEL);
  moe_dispatch<<<NEXP * CAP, 256, 0, stream>>>(x, slot2tok, disp);
  moe_gemm<0> <<<dim3(FFN / 64, CAP / 128, NEXP), 256, 0, stream>>>(
      disp, w1, b1, h, nullptr, nullptr, nullptr, CAP, FFN, DMODEL);
  moe_zero_dropped<<<S_TOK, 256, 0, stream>>>(tok_slot, out);
  moe_gemm<1> <<<dim3(DMODEL / 64, CAP / 128, NEXP), 256, 0, stream>>>(
      h, w2, b2, nullptr, slot2tok, tok_gate, out, CAP, DMODEL, FFN);
}
